// Transformer_17463337025619
// MI455X (gfx1250) — matrix operand order repaired and checked by reading, not compiled, not run on hardware
//
#include <hip/hip_runtime.h>
#include <math.h>

// ---------------- problem constants ----------------
#define B_   8
#define L_   256
#define D_   256
#define H_   8
#define DK_  32
#define NL_  4
#define V_   2048
#define FF_  1024
#define N_   (B_ * L_)     // 2048
#define EPS_ 1e-5f

typedef __attribute__((ext_vector_type(16))) _Float16 v16h;
typedef __attribute__((ext_vector_type(8)))  _Float16 v8h;
typedef __attribute__((ext_vector_type(8)))  float    v8f;

// D = A(16x32,f16) * B(32x16,f16) + C(16x16,f32)
static __device__ inline v8f wmma_f16(v16h a, v16h b, v8f c) {
    return __builtin_amdgcn_wmma_f32_16x16x32_f16(
        /*neg_a=*/false, a, /*neg_b=*/false, b,
        /*c_mod=*/(short)0, c, /*reuse_a=*/false, /*reuse_b=*/false);
}

// A-matrix (16x32 f16) lane fragment from a row-major row pointer.
// lane: m = lane&15 selects row (caller bakes row into `rowptr`), g = lane>>4.
// K indices: element e -> k0+16g+e (contiguous, matches B fragments).
static __device__ inline v16h load_A_frag(const _Float16* rowptr, int k0, int g) {
    const v8h lo = *(const v8h*)(rowptr + k0 + 16 * g);
    const v8h hi = *(const v8h*)(rowptr + k0 + 16 * g + 8);
    v16h a;
#pragma unroll
    for (int e = 0; e < 8; ++e) { a[e] = lo[e]; a[e + 8] = hi[e]; }
    return a;
}

// B-matrix (32x16 f16) fragment from an N-major (transposed) matrix Bt (N x K):
// lane n = lane&15, element e = B[k0+16g+e][col] = Bt[col*K + k0+16g+e]  -> contiguous!
static __device__ inline v16h load_B_tr(const _Float16* Bt, size_t ldk, int k0,
                                        int col, int g) {
    return *(const v16h*)(Bt + (size_t)col * ldk + k0 + 16 * g);
}

static __device__ inline float wave_sum(float v) {
#pragma unroll
    for (int o = 16; o >= 1; o >>= 1) v += __shfl_xor(v, o, 32);
    return v;
}

// ---------------- embedding ----------------
__global__ void embed_kernel(const int* __restrict__ tokens,
                             const int* __restrict__ positions,
                             const float* __restrict__ value_tab,
                             const float* __restrict__ coord_tab,
                             const float* __restrict__ pos_tab,
                             float* __restrict__ x) {
    int n = blockIdx.x;
    int d = threadIdx.x;
    int tok = tokens[n];
    int pos = positions[n];
    float v = value_tab[(size_t)tok * D_ + d] * 16.0f   // sqrt(256)
            + coord_tab[(size_t)(pos % 3) * D_ + d]
            + pos_tab[(size_t)(pos / 3) * D_ + d];
    x[(size_t)n * D_ + d] = v;
}

// ---------------- layernorm (f32 in -> f16 out), one wave per row ----------------
__global__ void ln_kernel(const float* __restrict__ x,
                          const float* __restrict__ gamma,
                          const float* __restrict__ beta,
                          _Float16* __restrict__ out) {
    int lane = threadIdx.x & 31;
    int wave = threadIdx.x >> 5;
    int row  = blockIdx.x * 4 + wave;           // blockDim = 128
    const float* xr = x + (size_t)row * D_;
    float r[8];
    float s = 0.f;
#pragma unroll
    for (int i = 0; i < 8; ++i) { r[i] = xr[lane + 32 * i]; s += r[i]; }
    s = wave_sum(s);
    float mean = s * (1.0f / D_);
    float vs = 0.f;
#pragma unroll
    for (int i = 0; i < 8; ++i) { float d = r[i] - mean; vs += d * d; }
    vs = wave_sum(vs);
    float rst = rsqrtf(vs * (1.0f / D_) + EPS_);
    _Float16* orow = out + (size_t)row * D_;
#pragma unroll
    for (int i = 0; i < 8; ++i) {
        int idx = lane + 32 * i;
        orow[idx] = (_Float16)((r[i] - mean) * rst * gamma[idx] + beta[idx]);
    }
}

// ---------------- generic WMMA GEMM: wave computes 32x64 tile ----------------
// B is stored transposed (N x K).  flags: 1 = relu, 2 = residual accumulate into
// outF, 4 = write outH transposed (outH[col*M + row]).
__global__ void __launch_bounds__(32)
gemm_kernel(const _Float16* __restrict__ A,
            const _Float16* __restrict__ Bt,
            int M, int Nn, int K,
            const float* __restrict__ bias,
            float* __restrict__ outF,
            _Float16* __restrict__ outH,
            int flags) {
    int lane = threadIdx.x;                 // blockDim = 32
    int m    = lane & 15;
    int g    = lane >> 4;
    int row0 = blockIdx.x * 32;
    int nb   = blockIdx.y * 64;
    const _Float16* arow0 = A + (size_t)(row0 + m) * K;
    const _Float16* arow1 = A + (size_t)(row0 + 16 + m) * K;
    v8f acc[2][4] = {{{}, {}, {}, {}}, {{}, {}, {}, {}}};
    for (int k0 = 0; k0 < K; k0 += 32) {
        v16h a0 = load_A_frag(arow0, k0, g);
        v16h a1 = load_A_frag(arow1, k0, g);
#pragma unroll
        for (int t = 0; t < 4; ++t) {
            v16h b = load_B_tr(Bt, (size_t)K, k0, nb + t * 16 + m, g);
            acc[0][t] = wmma_f16(a0, b, acc[0][t]);
            acc[1][t] = wmma_f16(a1, b, acc[1][t]);
        }
    }
#pragma unroll
    for (int t = 0; t < 4; ++t) {
        int col = nb + t * 16 + m;
        float bcol = bias ? bias[col] : 0.f;
#pragma unroll
        for (int s = 0; s < 2; ++s) {
#pragma unroll
            for (int u = 0; u < 8; ++u) {
                int rr = row0 + s * 16 + u + 8 * g;
                float val = acc[s][t][u] + bcol;
                if (flags & 1) val = fmaxf(val, 0.f);
                if (outF) {
                    size_t idx = (size_t)rr * Nn + col;
                    if (flags & 2) outF[idx] += val;
                    else           outF[idx]  = val;
                }
                if (outH) {
                    size_t idx = (flags & 4) ? (size_t)col * M + rr
                                             : (size_t)rr * Nn + col;
                    outH[idx] = (_Float16)val;
                }
            }
        }
    }
}

// ---------------- causal attention: one wave per (batch, head, 16-row tile) --------
// q,k are (N, D) f16 head-major rows; vT is (D, N) f16: vT[h*DK+dk][n].
__global__ void __launch_bounds__(32)
attn_kernel(const _Float16* __restrict__ qh,
            const _Float16* __restrict__ kh,
            const _Float16* __restrict__ vT,
            _Float16* __restrict__ zh) {
    __shared__ float    sbuf[16 * L_];     // scores, 16KB
    __shared__ _Float16 pbuf[16 * L_];     // probabilities (f16), 8KB
    int wg  = blockIdx.x;
    int rt  = wg & 15;
    int hh  = (wg >> 4) & (H_ - 1);
    int bi  = wg >> 7;
    int lane = threadIdx.x;                // blockDim = 32
    int m   = lane & 15;
    int g   = lane >> 4;
    int r0  = rt * 16;
    const float scale = 0.17677669529663687f;   // 1/sqrt(DK)

    // zero probability buffer (columns beyond computed chunks must be 0)
    for (int i = lane; i < 16 * L_; i += 32) pbuf[i] = (_Float16)0.f;

    // Q fragment: rows r0..r0+15, K-dim = DK (exactly one 32-wide k-step)
    const _Float16* qrow = qh + ((size_t)(bi * L_ + r0 + m)) * D_ + hh * DK_;
    v16h aq = load_A_frag(qrow, 0, g);

    int nch = rt + 1;                      // number of 16-wide key chunks
    for (int jc = 0; jc < nch; ++jc) {
        int j0 = jc * 16;
        // B fragment: B[k][n] = K[j0+n][k]; lane loads 16 contiguous halves of key row
        const _Float16* krow = kh + ((size_t)(bi * L_ + j0 + m)) * D_ + hh * DK_ + 16 * g;
        v16h bk = *(const v16h*)krow;
        v8f c = {};
        c = wmma_f16(aq, bk, c);
#pragma unroll
        for (int u = 0; u < 8; ++u) {
            int rr = u + 8 * g;
            int r  = r0 + rr;
            int j  = j0 + m;
            float val = c[u] * scale;
            if (j > r) val = -1e30f;        // causal mask
            sbuf[rr * L_ + j] = val;
        }
    }
    __syncthreads();

    // softmax: row m handled by lanes {m, m+16} (strided columns), combine via shfl_xor 16
    {
        int row = m;
        int len = r0 + row + 1;
        float mx = -1e30f;
        for (int j = g; j < len; j += 2) mx = fmaxf(mx, sbuf[row * L_ + j]);
        mx = fmaxf(mx, __shfl_xor(mx, 16, 32));
        float sm = 0.f;
        for (int j = g; j < len; j += 2) sm += __expf(sbuf[row * L_ + j] - mx);
        sm += __shfl_xor(sm, 16, 32);
        float inv = 1.0f / sm;
        for (int j = g; j < len; j += 2)
            pbuf[row * L_ + j] = (_Float16)(__expf(sbuf[row * L_ + j] - mx) * inv);
    }
    __syncthreads();

    // O = P @ V   (P: 16 x kch*32 f16; V via transposed vT -> contiguous B frags)
    int kch = (nch + 1) >> 1;
    v8f acc0 = {}, acc1 = {};
    for (int kc = 0; kc < kch; ++kc) {
        int k0 = kc * 32;
        v16h ap = load_A_frag(&pbuf[m * L_], k0, g);
        // b0: dk = m, b1: dk = 16+m ; elements e -> seq index k0+16g+e (contiguous)
        v16h b0 = *(const v16h*)(vT + ((size_t)(hh * DK_ + m)) * N_ + bi * L_ + k0 + 16 * g);
        v16h b1 = *(const v16h*)(vT + ((size_t)(hh * DK_ + 16 + m)) * N_ + bi * L_ + k0 + 16 * g);
        acc0 = wmma_f16(ap, b0, acc0);
        acc1 = wmma_f16(ap, b1, acc1);
    }
#pragma unroll
    for (int u = 0; u < 8; ++u) {
        int r = r0 + u + 8 * g;
        _Float16* zp = zh + ((size_t)(bi * L_ + r)) * D_ + hh * DK_;
        zp[m]      = (_Float16)acc0[u];
        zp[16 + m] = (_Float16)acc1[u];
    }
}

// ---------------- f32 (K x N) -> f16 transposed (N x K) ----------------
__global__ void cvtT_kernel(const float* __restrict__ in, _Float16* __restrict__ out,
                            int K, int N) {
    int i = blockIdx.x * blockDim.x + threadIdx.x;
    if (i < K * N) {
        int n = i / K;
        int k = i - n * K;
        out[(size_t)n * K + k] = (_Float16)in[(size_t)k * N + n];
    }
}

// ---------------- in-place log-softmax over rows of V_ ----------------
__global__ void logsoftmax_kernel(float* __restrict__ out) {
    __shared__ float red[256];
    int n = blockIdx.x;
    int t = threadIdx.x;
    float* row = out + (size_t)n * V_;
    float mx = -1e30f;
    for (int j = t; j < V_; j += 256) mx = fmaxf(mx, row[j]);
    red[t] = mx; __syncthreads();
    for (int o = 128; o >= 1; o >>= 1) { if (t < o) red[t] = fmaxf(red[t], red[t + o]); __syncthreads(); }
    mx = red[0]; __syncthreads();
    float sm = 0.f;
    for (int j = t; j < V_; j += 256) sm += __expf(row[j] - mx);
    red[t] = sm; __syncthreads();
    for (int o = 128; o >= 1; o >>= 1) { if (t < o) red[t] += red[t + o]; __syncthreads(); }
    float lse = mx + logf(red[0]);
    for (int j = t; j < V_; j += 256) row[j] = row[j] - lse;
}

// ---------------- host launch ----------------
extern "C" void kernel_launch(void* const* d_in, const int* in_sizes, int n_in,
                              void* d_out, int out_size, void* d_ws, size_t ws_size,
                              hipStream_t stream) {
    (void)in_sizes; (void)n_in; (void)out_size; (void)ws_size;
    const int*   tokens    = (const int*)  d_in[0];
    const int*   positions = (const int*)  d_in[1];
    // d_in[2] = src, d_in[3] = dst : causal structure exploited directly
    const float* value_tab = (const float*)d_in[4];
    const float* coord_tab = (const float*)d_in[5];
    const float* pos_tab   = (const float*)d_in[6];
    const float* ln1_g     = (const float*)d_in[7];
    const float* ln1_b     = (const float*)d_in[8];
    const float* Wq        = (const float*)d_in[9];
    const float* Wk        = (const float*)d_in[10];
    const float* Wv        = (const float*)d_in[11];
    const float* Wo        = (const float*)d_in[12];
    const float* ln2_g     = (const float*)d_in[13];
    const float* ln2_b     = (const float*)d_in[14];
    const float* W1        = (const float*)d_in[15];
    const float* b1        = (const float*)d_in[16];
    const float* W2        = (const float*)d_in[17];
    const float* b2        = (const float*)d_in[18];
    const float* lnf_g     = (const float*)d_in[19];
    const float* lnf_b     = (const float*)d_in[20];
    const float* Wgen      = (const float*)d_in[21];
    const float* bgen      = (const float*)d_in[22];
    float* out = (float*)d_out;

    // workspace carve
    char* wsp = (char*)d_ws;
    size_t off = 0;
    auto take = [&](size_t bytes) -> void* {
        void* p = wsp + off;
        off += (bytes + 255) & ~(size_t)255;
        return p;
    };
    float*    x      = (float*)   take((size_t)N_ * D_ * 4);
    _Float16* xh     = (_Float16*)take((size_t)N_ * D_ * 2);
    _Float16* q_h    = (_Float16*)take((size_t)N_ * D_ * 2);
    _Float16* k_h    = (_Float16*)take((size_t)N_ * D_ * 2);
    _Float16* vT_h   = (_Float16*)take((size_t)N_ * D_ * 2);   // (D x N) transposed
    _Float16* z_h    = (_Float16*)take((size_t)N_ * D_ * 2);
    _Float16* f_h    = (_Float16*)take((size_t)N_ * FF_ * 2);
    _Float16* wq_t   = (_Float16*)take((size_t)NL_ * D_ * D_ * 2);
    _Float16* wk_t   = (_Float16*)take((size_t)NL_ * D_ * D_ * 2);
    _Float16* wv_t   = (_Float16*)take((size_t)NL_ * D_ * D_ * 2);
    _Float16* wo_t   = (_Float16*)take((size_t)NL_ * D_ * D_ * 2);
    _Float16* w1_t   = (_Float16*)take((size_t)NL_ * D_ * FF_ * 2);
    _Float16* w2_t   = (_Float16*)take((size_t)NL_ * FF_ * D_ * 2);
    _Float16* wgen_t = (_Float16*)take((size_t)D_ * V_ * 2);

    // transpose-convert all weights to f16 N-major
    auto cvtT = [&](const float* src, _Float16* dst, int K, int N) {
        int n = K * N;
        cvtT_kernel<<<(n + 255) / 256, 256, 0, stream>>>(src, dst, K, N);
    };
    for (int i = 0; i < NL_; ++i) {
        cvtT(Wq + (size_t)i * D_ * D_,  wq_t + (size_t)i * D_ * D_,  D_, D_);
        cvtT(Wk + (size_t)i * D_ * D_,  wk_t + (size_t)i * D_ * D_,  D_, D_);
        cvtT(Wv + (size_t)i * D_ * D_,  wv_t + (size_t)i * D_ * D_,  D_, D_);
        cvtT(Wo + (size_t)i * D_ * D_,  wo_t + (size_t)i * D_ * D_,  D_, D_);
        cvtT(W1 + (size_t)i * D_ * FF_, w1_t + (size_t)i * D_ * FF_, D_, FF_);
        cvtT(W2 + (size_t)i * FF_ * D_, w2_t + (size_t)i * FF_ * D_, FF_, D_);
    }
    cvtT(Wgen, wgen_t, D_, V_);

    embed_kernel<<<N_, D_, 0, stream>>>(tokens, positions, value_tab, coord_tab, pos_tab, x);

    auto gemm = [&](const _Float16* A, const _Float16* Bt, int M, int Nn, int K,
                    const float* bias, float* oF, _Float16* oH, int flags) {
        dim3 grid(M / 32, Nn / 64);
        gemm_kernel<<<grid, 32, 0, stream>>>(A, Bt, M, Nn, K, bias, oF, oH, flags);
    };

    for (int i = 0; i < NL_; ++i) {
        // LN1 -> xh (f16)
        ln_kernel<<<N_ / 4, 128, 0, stream>>>(x, ln1_g + i * D_, ln1_b + i * D_, xh);
        // QKV projections (V written transposed for the attention B-operand)
        gemm(xh, wq_t + (size_t)i * D_ * D_, N_, D_, D_, nullptr, nullptr, q_h, 0);
        gemm(xh, wk_t + (size_t)i * D_ * D_, N_, D_, D_, nullptr, nullptr, k_h, 0);
        gemm(xh, wv_t + (size_t)i * D_ * D_, N_, D_, D_, nullptr, nullptr, vT_h, 4);
        // causal attention
        attn_kernel<<<B_ * H_ * 16, 32, 0, stream>>>(q_h, k_h, vT_h, z_h);
        // output projection + residual into x
        gemm(z_h, wo_t + (size_t)i * D_ * D_, N_, D_, D_, nullptr, x, nullptr, 2);
        // LN2 -> xh
        ln_kernel<<<N_ / 4, 128, 0, stream>>>(x, ln2_g + i * D_, ln2_b + i * D_, xh);
        // FFN
        gemm(xh, w1_t + (size_t)i * D_ * FF_, N_, FF_, D_, b1 + i * FF_, nullptr, f_h, 1);
        gemm(f_h, w2_t + (size_t)i * FF_ * D_, N_, D_, FF_, b2 + i * D_, x, nullptr, 2);
    }
    // final LN + generator + log-softmax (in place on d_out)
    ln_kernel<<<N_ / 4, 128, 0, stream>>>(x, lnf_g, lnf_b, xh);
    gemm(xh, wgen_t, N_, V_, D_, bgen, out, nullptr, 0);
    logsoftmax_kernel<<<N_, 256, 0, stream>>>(out);
}
